// RQSINN_79104707658011
// MI455X (gfx1250) — matrix operand order repaired and checked by reading, not compiled, not run on hardware
//
#include <hip/hip_runtime.h>
#include <hip/hip_bf16.h>

typedef __attribute__((ext_vector_type(16))) _Float16 v16h;
typedef __attribute__((ext_vector_type(8)))  float    v8f;

#define BATCH 65536
#define DIM   64
#define SPLIT 32
#define HID   128
#define PDIM  25
#define NB    8
#define NLAYERS 6
#define RMINF (-5.0f)
#define RMAXF (5.0f)
#define MIN_BIN 1e-4f
#define MIN_SLOPE 1e-4f

#define WAVES 4
#define ROWS_PER_BLOCK 64   // 4 waves * 16 rows

union ABu { v16h h; uint4 u[2]; };

// A operand (16xK tile, 16-bit): lane<16 -> row=lane, K = {kb..kb+7, kb+16..kb+23}
//                                lane>=16 -> row=lane-16, K = {kb+8..kb+15, kb+24..kb+31}
__device__ __forceinline__ v16h load_A_lds(const _Float16* base, int stride, int kbase, int lane) {
    int row = lane & 15, sel = lane >> 4;
    const _Float16* p0 = base + row * stride + kbase + sel * 8;
    ABu r;
    r.u[0] = *(const uint4*)(p0);
    r.u[1] = *(const uint4*)(p0 + 16);
    return r.h;
}

// B operand (Kx16 tile, 16-bit), weights stored row-major [N,K] (B[k][n] = W[n][k]):
// lane holds column n = nbase + (lane&15); K = {kb+sel*8..+7, kb+16+sel*8..+7}, sel = lane>>4
__device__ __forceinline__ v16h load_B_glb(const _Float16* w, int Ktot, int nbase, int kbase, int lane) {
    int col = lane & 15, sel = lane >> 4;
    const _Float16* p = w + (size_t)(nbase + col) * Ktot + kbase + sel * 8;
    ABu r;
    r.u[0] = *(const uint4*)(p);
    r.u[1] = *(const uint4*)(p + 16);
    return r.h;
}

__device__ __forceinline__ v8f wmma16(v16h a, v16h b, v8f c) {
    return __builtin_amdgcn_wmma_f32_16x16x32_f16(false, a, false, b, (short)0, c, false, false);
}

__global__ void f32_to_f16_kernel(const float* __restrict__ src, _Float16* __restrict__ dst, int n) {
    int i = blockIdx.x * blockDim.x + threadIdx.x;
    int stride = gridDim.x * blockDim.x;
    for (; i < n; i += stride) dst[i] = (_Float16)src[i];
}

__global__ __launch_bounds__(128)
void flow_layer_kernel(const float* __restrict__ xin, float* __restrict__ xout,
                       const _Float16* __restrict__ w1, const float* __restrict__ b1,
                       const _Float16* __restrict__ w2, const float* __restrict__ b2,
                       const _Float16* __restrict__ w3, const float* __restrict__ b3,
                       const _Float16* __restrict__ w4, const float* __restrict__ b4,
                       const int* __restrict__ midx, const int* __restrict__ tidx,
                       const float* __restrict__ ld_in, float* __restrict__ ld_out) {
    __shared__ __align__(16) _Float16 s_a[WAVES][16 * HID];
    __shared__ __align__(16) _Float16 s_b[WAVES][16 * HID];
    __shared__ __align__(16) float    s_par[WAVES][16 * 400];
    __shared__ int s_m[32], s_t[32];

    int tid = threadIdx.x;
    if (tid < 32)      s_m[tid]      = midx[tid];
    else if (tid < 64) s_t[tid - 32] = tidx[tid - 32];
    __syncthreads();

    int wave = tid >> 5;
    int lane = tid & 31;
    int sel  = lane >> 4;
    int ncol = lane & 15;
    int rowBase = blockIdx.x * ROWS_PER_BLOCK + wave * 16;

    _Float16* A  = s_a[wave];
    _Float16* Bf = s_b[wave];
    float*    SP = s_par[wave];

    // Stage xm (masked dims) as f16, 16 rows x 32 cols, stride 32
    #pragma unroll
    for (int rr = 0; rr < 16; ++rr)
        A[rr * SPLIT + lane] = (_Float16)xin[(size_t)(rowBase + rr) * DIM + s_m[lane]];

    // ---- GEMM1: h1 = relu(xm @ W1^T + b1)   [16x32] x [32x128] ----
    {
        v16h a = load_A_lds(A, SPLIT, 0, lane);
        #pragma unroll
        for (int nt = 0; nt < 8; ++nt) {
            int nb = nt * 16;
            v8f acc = {};
            acc = wmma16(a, load_B_glb(w1, SPLIT, nb, 0, lane), acc);
            float bias = b1[nb + ncol];
            #pragma unroll
            for (int j = 0; j < 8; ++j) {
                float v = acc[j] + bias; v = v > 0.f ? v : 0.f;
                Bf[(j + sel * 8) * HID + nb + ncol] = (_Float16)v;
            }
        }
    }
    // ---- GEMM2: h2 = relu(h1 @ W2^T + b2)   src Bf -> dst A ----
    {
        v16h a0 = load_A_lds(Bf, HID, 0, lane);
        v16h a1 = load_A_lds(Bf, HID, 32, lane);
        v16h a2 = load_A_lds(Bf, HID, 64, lane);
        v16h a3 = load_A_lds(Bf, HID, 96, lane);
        #pragma unroll
        for (int nt = 0; nt < 8; ++nt) {
            int nb = nt * 16;
            v8f acc = {};
            acc = wmma16(a0, load_B_glb(w2, HID, nb, 0, lane), acc);
            acc = wmma16(a1, load_B_glb(w2, HID, nb, 32, lane), acc);
            acc = wmma16(a2, load_B_glb(w2, HID, nb, 64, lane), acc);
            acc = wmma16(a3, load_B_glb(w2, HID, nb, 96, lane), acc);
            float bias = b2[nb + ncol];
            #pragma unroll
            for (int j = 0; j < 8; ++j) {
                float v = acc[j] + bias; v = v > 0.f ? v : 0.f;
                A[(j + sel * 8) * HID + nb + ncol] = (_Float16)v;
            }
        }
    }
    // ---- GEMM3: h3 = relu(h2 @ W3^T + b3)   src A -> dst Bf ----
    {
        v16h a0 = load_A_lds(A, HID, 0, lane);
        v16h a1 = load_A_lds(A, HID, 32, lane);
        v16h a2 = load_A_lds(A, HID, 64, lane);
        v16h a3 = load_A_lds(A, HID, 96, lane);
        #pragma unroll
        for (int nt = 0; nt < 8; ++nt) {
            int nb = nt * 16;
            v8f acc = {};
            acc = wmma16(a0, load_B_glb(w3, HID, nb, 0, lane), acc);
            acc = wmma16(a1, load_B_glb(w3, HID, nb, 32, lane), acc);
            acc = wmma16(a2, load_B_glb(w3, HID, nb, 64, lane), acc);
            acc = wmma16(a3, load_B_glb(w3, HID, nb, 96, lane), acc);
            float bias = b3[nb + ncol];
            #pragma unroll
            for (int j = 0; j < 8; ++j) {
                float v = acc[j] + bias; v = v > 0.f ? v : 0.f;
                Bf[(j + sel * 8) * HID + nb + ncol] = (_Float16)v;
            }
        }
    }
    // ---- GEMM4 + RQS spline, two 400-column passes ----
    v16h a0 = load_A_lds(Bf, HID, 0, lane);
    v16h a1 = load_A_lds(Bf, HID, 32, lane);
    v16h a2 = load_A_lds(Bf, HID, 64, lane);
    v16h a3 = load_A_lds(Bf, HID, 96, lane);

    int r = lane & 15;
    size_t growoff = (size_t)(rowBase + r) * DIM;
    float ldsum = 0.f;
    const float OFF = 0.54116774f;            // log(exp(1 - MIN_SLOPE) - 1)
    const float binscale = 10.f - NB * MIN_BIN;

    #pragma unroll
    for (int pass = 0; pass < 2; ++pass) {
        #pragma unroll
        for (int nt = 0; nt < 25; ++nt) {
            int nb = pass * 400 + nt * 16;
            v8f acc = {};
            acc = wmma16(a0, load_B_glb(w4, HID, nb, 0, lane), acc);
            acc = wmma16(a1, load_B_glb(w4, HID, nb, 32, lane), acc);
            acc = wmma16(a2, load_B_glb(w4, HID, nb, 64, lane), acc);
            acc = wmma16(a3, load_B_glb(w4, HID, nb, 96, lane), acc);
            float bias = b4[nb + ncol];
            #pragma unroll
            for (int j = 0; j < 8; ++j)
                SP[(j + sel * 8) * 400 + nt * 16 + ncol] = acc[j] + bias;
        }
        // each lane: row r, 8 transform dims  t = pass*16 + sel*8 + tt
        #pragma unroll
        for (int tt = 0; tt < 8; ++tt) {
            int tloc = sel * 8 + tt;
            int t = pass * 16 + tloc;
            const float* p = SP + r * 400 + tloc * 25;

            // pass-through masked dim (exact f32 copy from input)
            int mi = s_m[t];
            xout[growoff + mi] = xin[growoff + mi];

            float x = xin[growoff + s_t[t]];

            // widths softmax -> knot x positions
            float e0v[NB], e1v[NB], xpos[NB + 1], ypos[NB + 1], d[NB + 1];
            float mx = p[0];
            #pragma unroll
            for (int i = 1; i < NB; ++i) mx = fmaxf(mx, p[i]);
            float se = 0.f;
            #pragma unroll
            for (int i = 0; i < NB; ++i) { e0v[i] = __expf(p[i] - mx); se += e0v[i]; }
            float invw = binscale / se;
            xpos[0] = RMINF; float c = 0.f;
            #pragma unroll
            for (int i = 0; i < NB; ++i) { c += e0v[i] * invw + MIN_BIN; xpos[i + 1] = RMINF + c; }
            // heights softmax -> knot y positions
            float mh = p[NB];
            #pragma unroll
            for (int i = 1; i < NB; ++i) mh = fmaxf(mh, p[NB + i]);
            float sh = 0.f;
            #pragma unroll
            for (int i = 0; i < NB; ++i) { e1v[i] = __expf(p[NB + i] - mh); sh += e1v[i]; }
            float invh = binscale / sh;
            ypos[0] = RMINF; c = 0.f;
            #pragma unroll
            for (int i = 0; i < NB; ++i) { c += e1v[i] * invh + MIN_BIN; ypos[i + 1] = RMINF + c; }
            // slopes: softplus(raw + off) + min_slope
            #pragma unroll
            for (int i = 0; i <= NB; ++i) {
                float z = p[2 * NB + i] + OFF;
                float sp = (z > 15.f) ? z : __logf(1.f + __expf(z));
                d[i] = sp + MIN_SLOPE;
            }
            bool below = x < RMINF, above = x > RMAXF;
            int bi = 0;
            #pragma unroll
            for (int i = 1; i < NB; ++i) bi += (xpos[i] <= x) ? 1 : 0;
            if (bi > NB - 1) bi = NB - 1;
            float xk = xpos[0], xk1 = xpos[1], yk = ypos[0], yk1 = ypos[1], dk = d[0], dk1 = d[1];
            #pragma unroll
            for (int i = 1; i < NB; ++i) {
                if (bi == i) { xk = xpos[i]; xk1 = xpos[i + 1]; yk = ypos[i]; yk1 = ypos[i + 1]; dk = d[i]; dk1 = d[i + 1]; }
            }
            float w_ = xk1 - xk, h_ = yk1 - yk;
            float xi = (x - xk) / w_;
            xi = fminf(fmaxf(xi, 0.f), 1.f);
            float s_ = h_ / w_;
            float xim = 1.f - xi, q = xi * xim;
            float num = s_ * xi * xi + dk * q;
            float den = s_ + (dk1 + dk - 2.f * s_) * q;
            float ysp = yk + h_ * num / den;
            float dd = s_ * s_ * (dk1 * xi * xi + 2.f * s_ * q + dk * xim * xim) / (den * den);
            float yl = (x - RMINF) * d[0] + RMINF;
            float yr = (x - RMAXF) * d[NB] + RMAXF;
            float y = below ? yl : (above ? yr : ysp);
            float ldv = below ? __logf(d[0]) : (above ? __logf(d[NB]) : __logf(dd));
            xout[growoff + s_t[t]] = y;
            ldsum += ldv;
        }
    }
    // combine the two t-halves (lane and lane+16 hold same row) and write logdet
    ldsum += __shfl_xor(ldsum, 16, 32);
    if (lane < 16) {
        float base = ld_in ? ld_in[rowBase + r] : 0.f;
        ld_out[rowBase + r] = base + ldsum;
    }
}

extern "C" void kernel_launch(void* const* d_in, const int* in_sizes, int n_in,
                              void* d_out, int out_size, void* d_ws, size_t ws_size,
                              hipStream_t stream) {
    (void)in_sizes; (void)n_in; (void)out_size; (void)ws_size;
    const float* x  = (const float*)d_in[0];
    const float* W1 = (const float*)d_in[1];
    const float* B1 = (const float*)d_in[2];
    const float* W2 = (const float*)d_in[3];
    const float* B2 = (const float*)d_in[4];
    const float* W3 = (const float*)d_in[5];
    const float* B3 = (const float*)d_in[6];
    const float* W4 = (const float*)d_in[7];
    const float* B4 = (const float*)d_in[8];
    const int*   MI = (const int*)d_in[9];
    const int*   TI = (const int*)d_in[10];

    char* ws = (char*)d_ws;
    size_t off = 0;
    _Float16* w1h = (_Float16*)(ws + off); off += (size_t)NLAYERS * HID * SPLIT * 2;
    _Float16* w2h = (_Float16*)(ws + off); off += (size_t)NLAYERS * HID * HID * 2;
    _Float16* w3h = (_Float16*)(ws + off); off += (size_t)NLAYERS * HID * HID * 2;
    _Float16* w4h = (_Float16*)(ws + off); off += (size_t)NLAYERS * SPLIT * PDIM * HID * 2;
    off = (off + 255) & ~(size_t)255;
    float* bufA  = (float*)(ws + off); off += (size_t)BATCH * DIM * 4;
    float* bufB  = (float*)(ws + off); off += (size_t)BATCH * DIM * 4;
    float* ldbuf = (float*)(ws + off); off += (size_t)BATCH * 4;

    const int cb = 256;
    f32_to_f16_kernel<<<(NLAYERS * HID * SPLIT + cb - 1) / cb, cb, 0, stream>>>(W1, w1h, NLAYERS * HID * SPLIT);
    f32_to_f16_kernel<<<(NLAYERS * HID * HID + cb - 1) / cb, cb, 0, stream>>>(W2, w2h, NLAYERS * HID * HID);
    f32_to_f16_kernel<<<(NLAYERS * HID * HID + cb - 1) / cb, cb, 0, stream>>>(W3, w3h, NLAYERS * HID * HID);
    f32_to_f16_kernel<<<(NLAYERS * SPLIT * PDIM * HID + cb - 1) / cb, cb, 0, stream>>>(W4, w4h, NLAYERS * SPLIT * PDIM * HID);

    float* yout = (float*)d_out;
    float* ldfinal = yout + (size_t)BATCH * DIM;

    const float* cur = x;
    float* bufs[2] = { bufA, bufB };
    dim3 grid(BATCH / ROWS_PER_BLOCK), block(128);
    for (int l = 0; l < NLAYERS; ++l) {
        float* out = (l == NLAYERS - 1) ? yout : bufs[l & 1];
        const float* ldin = (l == 0) ? nullptr : ldbuf;
        float* ldo = (l == NLAYERS - 1) ? ldfinal : ldbuf;
        flow_layer_kernel<<<grid, block, 0, stream>>>(
            cur, out,
            w1h + (size_t)l * HID * SPLIT,      B1 + (size_t)l * HID,
            w2h + (size_t)l * HID * HID,        B2 + (size_t)l * HID,
            w3h + (size_t)l * HID * HID,        B3 + (size_t)l * HID,
            w4h + (size_t)l * SPLIT * PDIM * HID, B4 + (size_t)l * SPLIT * PDIM,
            MI + l * SPLIT, TI + l * SPLIT,
            ldin, ldo);
        cur = out;
    }
}
